// RecurrentEmbeddingModel_23802708754579
// MI455X (gfx1250) — matrix operand order repaired and checked by reading, not compiled, not run on hardware
//
#include <hip/hip_runtime.h>
#include <hip/hip_bf16.h>

typedef __attribute__((ext_vector_type(16))) _Float16 v16h;
typedef __attribute__((ext_vector_type(2)))  __fp16   fp16x2;
typedef __attribute__((ext_vector_type(8)))  float    v8f;
typedef __attribute__((ext_vector_type(8)))  unsigned v8u;

namespace {
constexpr int kN     = 16384;
constexpr int kLanes = 2;
constexpr int kB     = kN * kLanes;   // 32768 RNN instances
constexpr int kH     = 32;            // hidden size
constexpr int kWidth = 513;           // MAXH + 1 (row stride of obs)
}

// Branch-free tanh. gfx1250 has a hardware tanh transcendental.
static __device__ __forceinline__ float fast_tanh(float x) {
#if __has_builtin(__builtin_amdgcn_tanhf)
  return __builtin_amdgcn_tanhf(x);
#else
  float xc = fminf(fmaxf(x, -10.0f), 10.0f);
  float y  = __builtin_amdgcn_exp2f(2.8853900817779268f * xc);  // 2*log2(e)*x
  return fmaf(-2.0f, __builtin_amdgcn_rcpf(y + 1.0f), 1.0f);
#endif
}

// Swap a dword between lanes L and L^16 (wave32) as a single VALU op.
static __device__ __forceinline__ unsigned xor16_swap_u32(unsigned v) {
#if __has_builtin(__builtin_amdgcn_permlanex16)
  return (unsigned)__builtin_amdgcn_permlanex16(
      (int)v, (int)v, 0x76543210, 0xfedcba98, false, false);
#else
  return (unsigned)__shfl_xor((int)v, 16, 32);
#endif
}

// Pack two f32 into a packed-f16 dword (v_cvt_pkrtz_f16_f32).
static __device__ __forceinline__ unsigned pk_f16(float a, float b) {
  fp16x2 p = __builtin_amdgcn_cvt_pkrtz(a, b);
  return __builtin_bit_cast(unsigned, p);
}

// ---------------------------------------------------------------------------
// Stage 1: batched tanh-RNN recurrence via WMMA.
// Each wave owns 16 batch columns. State h^T (32 x 16) lives as the WMMA
// B operand (f16). Per step:
//   D0 (j=0..15)  = wmma(A0 = w_hh^T[0:16,:],  B = h^T, C0)
//   D1 (j=16..31) = wmma(A1 = w_hh^T[16:32,:], B = h^T, C1)
//   h_new = tanh(D);  emb captured (packed f16) where t == len-1;  B rebuilt
//   from D with one lane-xor-16 permlane per packed dword pair.
// ---------------------------------------------------------------------------
__global__ __launch_bounds__(256) void rnn_wmma_kernel(
    const float* __restrict__ obs,    // [B, 513]; [:,0] = length
    const float* __restrict__ w_ih,   // [32]
    const float* __restrict__ w_hh,   // [32, 32] row-major (k, j)
    const float* __restrict__ b_ih,   // [32]
    const float* __restrict__ b_hh,   // [32]
    float* __restrict__ emb_ws)       // [B, 32] captured hidden states
{
  const int lane = threadIdx.x & 31;
  const int wave = (blockIdx.x * blockDim.x + threadIdx.x) >> 5;
  const int b0   = wave * 16;
  if (b0 >= kB) return;
  const int  col = lane & 15;
  const bool hi  = (lane & 16) != 0;
  const int  b   = b0 + col;

  // ---- Preload A = w_hh^T tiles in a contiguous-k A (16x32) order (same k order as B).
  // lane<16: e=0..15 -> K=e; lane>=16: +16 on K.
  // A[m][k] = w_hh[k][m]  (h_new[j] = sum_k h[k] * w_hh[k, j])
  v16h a0, a1;
#pragma unroll
  for (int e = 0; e < 16; ++e) {
    const int K = e + (hi ? 16 : 0);
    a0[e] = (_Float16)w_hh[K * kH + col];         // rows j = 0..15
    a1[e] = (_Float16)w_hh[K * kH + 16 + col];    // rows j = 16..31
  }

  // ---- Per-lane input weight / bias in the C/D (16x16 f32) layout:
  // vgpr r, lane<16 -> j = r (+tile*16); lane>=16 -> j = r+8 (+tile*16).
  float wih0[8], wih1[8], bias0[8], bias1[8];
#pragma unroll
  for (int r = 0; r < 8; ++r) {
    const int j0 = r + (hi ? 8 : 0);
    const int j1 = j0 + 16;
    wih0[r] = w_ih[j0]; bias0[r] = b_ih[j0] + b_hh[j0];
    wih1[r] = w_ih[j1]; bias1[r] = b_ih[j1] + b_hh[j1];
  }

  const float* __restrict__ rowp = obs + (long)b * kWidth;
  const int len   = (int)rowp[0];           // obs[:, :, 0] holds length
  const int lenm1 = len - 1;                // -1 when len==0 -> never captured

  // Wave-wide max length over the 16 columns (halves are duplicates).
  int ml = len;
#pragma unroll
  for (int off = 8; off > 0; off >>= 1) ml = max(ml, __shfl_xor(ml, off, 32));
  const int maxlen = __builtin_amdgcn_readfirstlane(ml);

  v8u bmu = {};            // h_prev^T = 0, packed f16 B operand as dwords
  v8u epk = {};            // captured emb, packed f16: [0..3]=tile0, [4..7]=tile1

  // Double-buffered input prefetch: 8 steps per chunk. Only the chunk BASE
  // is clamped: a chunk is consumed only when tc+8 <= 504 (loop bound), so
  // every consumed chunk has an exact base; the final (dead) prefetch is
  // clamped to stay in-bounds.
  float xcur[8], xnxt[8];
#pragma unroll
  for (int q = 0; q < 8; ++q) xcur[q] = rowp[q];

  for (int tc = 0; tc < maxlen; tc += 8) {
    const int pb = (tc + 8) <= (kWidth - 8) ? (tc + 8) : (kWidth - 8);
    const float* __restrict__ pp = rowp + pb;
#pragma unroll
    for (int q = 0; q < 8; ++q) xnxt[q] = pp[q];

#pragma unroll
    for (int q = 0; q < 8; ++q) {
      const int t = tc + q;
      const v16h bmat = __builtin_bit_cast(v16h, bmu);

      // C tiles: x_t * w_ih[j] + bias[j]
      v8f c0, c1;
#pragma unroll
      for (int r = 0; r < 8; ++r) {
        c0[r] = fmaf(xcur[q], wih0[r], bias0[r]);
        c1[r] = fmaf(xcur[q], wih1[r], bias1[r]);
      }
      v8f d0 = __builtin_amdgcn_wmma_f32_16x16x32_f16(
          false, a0, false, bmat, (short)0, c0, false, false);
      v8f d1 = __builtin_amdgcn_wmma_f32_16x16x32_f16(
          false, a1, false, bmat, (short)0, c1, false, false);

      float h0[8], h1[8];
#pragma unroll
      for (int r = 0; r < 8; ++r) {
        h0[r] = fast_tanh(d0[r]);
        h1[r] = fast_tanh(d1[r]);
      }

      const bool cap = (t == lenm1);   // last valid step for this column

      // Rebuild packed B operand (h_new^T). Dword s holds B elements
      // (2s, 2s+1). Low half (K base): lo lanes keep own h0 pack; hi lanes
      // need h1 pack from lane-16. High half: lo lanes need h0 pack from
      // lane+16; hi lanes keep own h1 pack. One permlanex16 per dword pair.
      // The same packs feed the (packed f16) emb capture.
#pragma unroll
      for (int s = 0; s < 4; ++s) {
        const unsigned p0 = pk_f16(h0[2 * s], h0[2 * s + 1]);
        const unsigned p1 = pk_f16(h1[2 * s], h1[2 * s + 1]);
        epk[s]     = cap ? p0 : epk[s];
        epk[4 + s] = cap ? p1 : epk[4 + s];
        const unsigned u = hi ? p0 : p1;
        const unsigned w = xor16_swap_u32(u);
        bmu[s]     = hi ? w  : p0;
        bmu[4 + s] = hi ? p1 : w;
      }
    }
#pragma unroll
    for (int q = 0; q < 8; ++q) xcur[q] = xnxt[q];
  }

  // ---- Unpack and write captured emb as [B, 32] (== feat [N, 64] row-major).
#pragma unroll
  for (int s = 0; s < 4; ++s) {
    const fp16x2 q0 = __builtin_bit_cast(fp16x2, epk[s]);
    const fp16x2 q1 = __builtin_bit_cast(fp16x2, epk[4 + s]);
    const int j = 2 * s + (hi ? 8 : 0);
    emb_ws[b * kH + j]          = (float)q0.x;
    emb_ws[b * kH + j + 1]      = (float)q0.y;
    emb_ws[b * kH + j + 16]     = (float)q1.x;
    emb_ws[b * kH + j + 17]     = (float)q1.y;
  }
}

// ---------------------------------------------------------------------------
// Stage 2: tiny MLP 64 -> 32 -> 32 -> 1 per N row. Weights staged in LDS,
// one thread per output row; ~100 MFLOP total, bandwidth-trivial.
// ---------------------------------------------------------------------------
__global__ __launch_bounds__(256) void mlp_kernel(
    const float* __restrict__ feat,   // [N, 64] (= emb_ws)
    const float* __restrict__ w1, const float* __restrict__ b1,
    const float* __restrict__ w2, const float* __restrict__ b2,
    const float* __restrict__ w3, const float* __restrict__ b3,
    float* __restrict__ out, int n_rows)
{
  __shared__ float sw1[64 * 32];
  __shared__ float sw2[32 * 32];
  __shared__ float sw3[32];
  __shared__ float sb1[32];
  __shared__ float sb2[32];
  for (int i = threadIdx.x; i < 64 * 32; i += blockDim.x) sw1[i] = w1[i];
  for (int i = threadIdx.x; i < 32 * 32; i += blockDim.x) sw2[i] = w2[i];
  if (threadIdx.x < 32) {
    sw3[threadIdx.x] = w3[threadIdx.x];
    sb1[threadIdx.x] = b1[threadIdx.x];
    sb2[threadIdx.x] = b2[threadIdx.x];
  }
  __syncthreads();

  const int n = blockIdx.x * blockDim.x + threadIdx.x;
  if (n >= n_rows) return;

  float acc[32];
#pragma unroll
  for (int k = 0; k < 32; ++k) acc[k] = sb1[k];
  for (int i = 0; i < 64; ++i) {
    const float f = feat[n * 64 + i];
#pragma unroll
    for (int k = 0; k < 32; ++k) acc[k] = fmaf(f, sw1[i * 32 + k], acc[k]);
  }
#pragma unroll
  for (int k = 0; k < 32; ++k) acc[k] = acc[k] > 0.0f ? acc[k] : 0.0f;

  float acc2[32];
#pragma unroll
  for (int k = 0; k < 32; ++k) acc2[k] = sb2[k];
  for (int i = 0; i < 32; ++i) {
    const float f = acc[i];
#pragma unroll
    for (int k = 0; k < 32; ++k) acc2[k] = fmaf(f, sw2[i * 32 + k], acc2[k]);
  }

  float o = b3[0];
#pragma unroll
  for (int k = 0; k < 32; ++k)
    o = fmaf(acc2[k] > 0.0f ? acc2[k] : 0.0f, sw3[k], o);

  out[n] = o;
}

extern "C" void kernel_launch(void* const* d_in, const int* in_sizes, int n_in,
                              void* d_out, int out_size, void* d_ws, size_t ws_size,
                              hipStream_t stream) {
  const float* obs  = (const float*)d_in[0];
  const float* w_ih = (const float*)d_in[1];
  const float* w_hh = (const float*)d_in[2];
  const float* b_ih = (const float*)d_in[3];
  const float* b_hh = (const float*)d_in[4];
  const float* w1   = (const float*)d_in[5];
  const float* b1   = (const float*)d_in[6];
  const float* w2   = (const float*)d_in[7];
  const float* b2   = (const float*)d_in[8];
  const float* w3   = (const float*)d_in[9];
  const float* b3   = (const float*)d_in[10];
  float* out    = (float*)d_out;
  float* emb_ws = (float*)d_ws;   // kB * kH floats = 4 MB

  // 2048 waves, one 16-column tile each; 8 waves / block.
  const int waves  = kB / 16;                 // 2048
  const int blocks = waves / 8;               // 256 blocks of 256 threads
  rnn_wmma_kernel<<<blocks, 256, 0, stream>>>(obs, w_ih, w_hh, b_ih, b_hh, emb_ws);

  const int mblocks = (kN + 255) / 256;
  mlp_kernel<<<mblocks, 256, 0, stream>>>(emb_ws, w1, b1, w2, b2, w3, b3, out, kN);
}
